// GraphAttentionEmbedding_47725676593772
// MI455X (gfx1250) — compile-verified, matrix operand order checked by reading, not run on hardware
//
#include <hip/hip_runtime.h>

// ---------------------------------------------------------------------------
// Problem constants (from reference)
// ---------------------------------------------------------------------------
#define NN     100000
#define EEDGES 640000
#define INF_   100
#define OUTF   128
#define KPAD   128
#define ECOLS  118      // edge_attr columns
#define EDIMK  125      // gathered edge feature dim (3 + 114 + 8)
#define DHEAD  32
#define SCALE_ 0.17677669529663687f  // 1/sqrt(32)

typedef __attribute__((ext_vector_type(16))) __bf16 v16bf;
typedef __attribute__((ext_vector_type(8)))  __bf16 v8bf;
typedef __attribute__((ext_vector_type(8)))  float  v8f;

// float -> bf16 with round-to-nearest-even (pack passes only)
static __device__ __forceinline__ __bf16 f2bf(float f) {
  unsigned u = __float_as_uint(f);
  unsigned r = u + 0x7FFFu + ((u >> 16) & 1u);
  unsigned short h = (unsigned short)(r >> 16);
  return __builtin_bit_cast(__bf16, h);
}

// Load a 16-element bf16 fragment as two 16-byte halves (both 16B aligned).
static __device__ __forceinline__ v16bf load_frag(const __bf16* __restrict__ base,
                                                  int off0, int off1) {
  v8bf lo = *(const v8bf*)(base + off0);
  v8bf hi = *(const v8bf*)(base + off1);
  return __builtin_shufflevector(lo, hi, 0, 1, 2, 3, 4, 5, 6, 7,
                                 8, 9, 10, 11, 12, 13, 14, 15);
}

// ordered-uint mapping for float atomic max
static __device__ __forceinline__ unsigned fmap(float f) {
  unsigned u = __float_as_uint(f);
  return (u & 0x80000000u) ? ~u : (u | 0x80000000u);
}
static __device__ __forceinline__ float funmap(unsigned u) {
  unsigned v = (u & 0x80000000u) ? (u & 0x7FFFFFFFu) : ~u;
  return __uint_as_float(v);
}

// ---------------------------------------------------------------------------
// Pack kernels: convert f32 operands to zero-padded bf16 layouts.
// ---------------------------------------------------------------------------
__global__ void pack_x_kernel(const float* __restrict__ X, __bf16* __restrict__ Xbf,
                              long nElem, int F) {
  long idx = (long)blockIdx.x * blockDim.x + threadIdx.x;
  if (idx >= nElem) return;
  long row = idx >> 7;
  int  k   = (int)(idx & 127);
  float v = (k < F) ? X[row * F + k] : 0.f;
  Xbf[idx] = f2bf(v);
}

// WT[c*128 + k] = W[k*128 + c]  (transposed, K padded to 128)
__global__ void pack_wT_kernel(const float* __restrict__ W, __bf16* __restrict__ WT,
                               int F) {
  int idx = blockIdx.x * blockDim.x + threadIdx.x;
  if (idx >= KPAD * OUTF) return;
  int c = idx >> 7;
  int k = idx & 127;
  float v = (k < F) ? W[k * OUTF + c] : 0.f;
  WT[idx] = f2bf(v);
}

static __device__ __forceinline__ float fetch_edge_feat(
    const float* __restrict__ ea, const float* __restrict__ emb, long e, int k) {
  if (k < 3)     return ea[e * ECOLS + k];
  if (k < 117)   return ea[e * ECOLS + k + 1];
  int et = (int)ea[e * ECOLS + 3];
  return emb[et * 8 + (k - 117)];
}

__global__ void pack_edge_kernel(const float* __restrict__ ea,
                                 const float* __restrict__ emb,
                                 __bf16* __restrict__ Ebf, long nElem) {
  long idx = (long)blockIdx.x * blockDim.x + threadIdx.x;
  if (idx >= nElem) return;
  long e = idx >> 7;
  int  k = (int)(idx & 127);
  float v = (k < EDIMK) ? fetch_edge_feat(ea, emb, e, k) : 0.f;
  Ebf[idx] = f2bf(v);
}

// ---------------------------------------------------------------------------
// Kernel: [Q|K|V|S] = X @ [Wq Wk Wv Ws] + bias   (bf16 WMMA, f32 accum)
// One wave = 16 rows x 128 cols of one weight matrix; 4 waves/block = 4 mats.
// Branch-free: N % 16 == 0, K padded to 128.
// ---------------------------------------------------------------------------
__global__ __launch_bounds__(128) void node_gemm_kernel(
    const __bf16* __restrict__ Xbf, const __bf16* __restrict__ WT4,
    const float* __restrict__ Bq, const float* __restrict__ Bk,
    const float* __restrict__ Bv, const float* __restrict__ Bs,
    float* __restrict__ out /* N x 512 */) {
  const int wave = threadIdx.x >> 5;
  const int lane = threadIdx.x & 31;
  const int row0 = blockIdx.x << 4;

  const __bf16* WT = WT4 + (long)wave * (KPAD * OUTF);
  const float*  B  = (wave == 0) ? Bq : (wave == 1) ? Bk : (wave == 2) ? Bv : Bs;

  const int r  = lane & 15;
  const int kb = (lane >> 4) << 3;
  const __bf16* arow = Xbf + (long)(row0 + r) * KPAD;

  v8f acc[8] = {};

#pragma unroll
  for (int ks = 0; ks < 4; ++ks) {
    const int o0 = (ks << 5) + kb;
    const int o1 = o0 + 16;
    v16bf a = load_frag(arow, o0, o1);
#pragma unroll
    for (int t = 0; t < 8; ++t) {
      const __bf16* bcol = WT + (long)((t << 4) + r) * KPAD;
      v16bf b = load_frag(bcol, o0, o1);
      acc[t] = __builtin_amdgcn_wmma_f32_16x16x32_bf16(
          false, a, false, b, (short)0, acc[t], false, false);
    }
  }

  const int mb = (lane >> 4) << 3;
#pragma unroll
  for (int t = 0; t < 8; ++t) {
    const int c0 = t << 4;
    const float bias = B[c0 + r];
#pragma unroll
    for (int i = 0; i < 8; ++i) {
      const long m = row0 + mb + i;
      out[m * 512 + (wave << 7) + c0 + r] = acc[t][i] + bias;
    }
  }
}

// ---------------------------------------------------------------------------
// Kernel: EE = Ebf @ We   (bf16 WMMA, f32 accum). One wave = 16 edges x 128.
// Branch-free: E % 64 == 0, K padded to 128.
// ---------------------------------------------------------------------------
__global__ __launch_bounds__(128) void edge_gemm_kernel(
    const __bf16* __restrict__ Ebf, const __bf16* __restrict__ WeT,
    float* __restrict__ EE) {
  const int wave = threadIdx.x >> 5;
  const int lane = threadIdx.x & 31;
  const long e0 = (((long)blockIdx.x << 2) + wave) << 4;

  const int r  = lane & 15;
  const int kb = (lane >> 4) << 3;
  const __bf16* arow = Ebf + (e0 + r) * KPAD;

  v8f acc[8] = {};

#pragma unroll
  for (int ks = 0; ks < 4; ++ks) {
    const int o0 = (ks << 5) + kb;
    const int o1 = o0 + 16;
    v16bf a = load_frag(arow, o0, o1);
#pragma unroll
    for (int t = 0; t < 8; ++t) {
      const __bf16* bcol = WeT + (long)((t << 4) + r) * KPAD;
      v16bf b = load_frag(bcol, o0, o1);
      acc[t] = __builtin_amdgcn_wmma_f32_16x16x32_bf16(
          false, a, false, b, (short)0, acc[t], false, false);
    }
  }

  const int mb = (lane >> 4) << 3;
#pragma unroll
  for (int t = 0; t < 8; ++t) {
    const int c0 = t << 4;
#pragma unroll
    for (int i = 0; i < 8; ++i) {
      EE[(e0 + mb + i) * OUTF + c0 + r] = acc[t][i];
    }
  }
}

// ---------------------------------------------------------------------------
// Kernel: alpha[e,h] = SCALE * dot(q[dst], k[src]+ee); atomic segment max.
// ---------------------------------------------------------------------------
__global__ void edge_alpha_kernel(
    const float* __restrict__ QKVS, const float* __restrict__ EE,
    const int* __restrict__ src, const int* __restrict__ dst,
    float* __restrict__ alpha, unsigned* __restrict__ mmax, long nEH) {
  long idx = (long)blockIdx.x * blockDim.x + threadIdx.x;
  if (idx >= nEH) return;
  const long e = idx >> 2;
  const int  h = (int)(idx & 3);
  const int  s = src[e];
  const int  d = dst[e];
  const float4* q  = (const float4*)(QKVS + (long)d * 512 + h * DHEAD);
  const float4* k  = (const float4*)(QKVS + (long)s * 512 + 128 + h * DHEAD);
  const float4* ee = (const float4*)(EE + e * OUTF + h * DHEAD);
  float acc = 0.f;
#pragma unroll
  for (int i = 0; i < 8; ++i) {
    float4 qv = q[i], kv = k[i], ev = ee[i];
    acc += qv.x * (kv.x + ev.x) + qv.y * (kv.y + ev.y) +
           qv.z * (kv.z + ev.z) + qv.w * (kv.w + ev.w);
  }
  acc *= SCALE_;
  alpha[idx] = acc;
  atomicMax(mmax + (long)d * 4 + h, fmap(acc));
}

// ---------------------------------------------------------------------------
// Kernel: ex = exp(alpha - max[dst]); atomic segment sum. (alpha in-place)
// ---------------------------------------------------------------------------
__global__ void edge_expsum_kernel(
    const int* __restrict__ dst, const unsigned* __restrict__ mmax,
    float* __restrict__ alpha, float* __restrict__ ssum, long nEH) {
  long idx = (long)blockIdx.x * blockDim.x + threadIdx.x;
  if (idx >= nEH) return;
  const long e = idx >> 2;
  const int  h = (int)(idx & 3);
  const int  d = dst[e];
  const float m  = funmap(mmax[(long)d * 4 + h]);
  const float ex = __expf(alpha[idx] - m);
  alpha[idx] = ex;
  atomicAdd(ssum + (long)d * 4 + h, ex);
}

// ---------------------------------------------------------------------------
// Kernel: msg = (v[src] + ee) * softmax; scatter-add into accum[dst].
// ---------------------------------------------------------------------------
__global__ void edge_scatter_kernel(
    const float* __restrict__ QKVS, const float* __restrict__ EE,
    const int* __restrict__ src, const int* __restrict__ dst,
    const float* __restrict__ ex, const float* __restrict__ ssum,
    float* __restrict__ accum, long nEC) {
  long idx = (long)blockIdx.x * blockDim.x + threadIdx.x;
  if (idx >= nEC) return;
  const long e = idx >> 7;
  const int  i = (int)(idx & 127);
  const int  h = i >> 5;
  const int  s = src[e];
  const int  d = dst[e];
  const float a = ex[e * 4 + h] / (ssum[(long)d * 4 + h] + 1e-16f);
  const float v = QKVS[(long)s * 512 + 256 + i];
  atomicAdd(accum + (long)d * OUTF + i, (v + EE[e * OUTF + i]) * a);
}

// ---------------------------------------------------------------------------
// Kernel: out = [relu](accum + skip)   (skip = x@Ws + bs at QKVS offset 384)
// ---------------------------------------------------------------------------
__global__ void finalize_kernel(
    const float* __restrict__ accum, const float* __restrict__ QKVS,
    float* __restrict__ out, long nNC, int do_relu) {
  long idx = (long)blockIdx.x * blockDim.x + threadIdx.x;
  if (idx >= nNC) return;
  const long row = idx >> 7;
  const int  col = (int)(idx & 127);
  float v = accum[idx] + QKVS[row * 512 + 384 + col];
  if (do_relu) v = fmaxf(v, 0.f);
  out[idx] = v;
}

// ---------------------------------------------------------------------------
// Launcher
// ---------------------------------------------------------------------------
extern "C" void kernel_launch(void* const* d_in, const int* in_sizes, int n_in,
                              void* d_out, int out_size, void* d_ws, size_t ws_size,
                              hipStream_t stream) {
  (void)in_sizes; (void)n_in; (void)out_size; (void)ws_size;
  const int N = NN, E = EEDGES;

  const float* x    = (const float*)d_in[0];
  const int*   eidx = (const int*)d_in[1];
  const int*   src  = eidx;
  const int*   dst  = eidx + E;
  const float* ea   = (const float*)d_in[2];
  const float* emb  = (const float*)d_in[3];

  const float* W[2][5];  // q,k,v,e,s
  const float* B[2][4];  // q,k,v,s
  for (int l = 0; l < 2; ++l) {
    int base = 4 + l * 9;
    W[l][0] = (const float*)d_in[base + 0]; B[l][0] = (const float*)d_in[base + 1];
    W[l][1] = (const float*)d_in[base + 2]; B[l][1] = (const float*)d_in[base + 3];
    W[l][2] = (const float*)d_in[base + 4]; B[l][2] = (const float*)d_in[base + 5];
    W[l][3] = (const float*)d_in[base + 6];                         // we
    W[l][4] = (const float*)d_in[base + 7]; B[l][3] = (const float*)d_in[base + 8];
  }

  // workspace layout
  char* ws = (char*)d_ws;
  float*    QKVS  = (float*)ws;    ws += (size_t)N * 512 * sizeof(float);
  float*    EE    = (float*)ws;    ws += (size_t)E * OUTF * sizeof(float);
  float*    accum = (float*)ws;    ws += (size_t)N * OUTF * sizeof(float);
  float*    alpha = (float*)ws;    ws += (size_t)E * 4 * sizeof(float);
  unsigned* mmax  = (unsigned*)ws; ws += (size_t)N * 4 * sizeof(unsigned);
  float*    ssum  = (float*)ws;    ws += (size_t)N * 4 * sizeof(float);
  float*    Hbuf  = (float*)ws;    ws += (size_t)N * OUTF * sizeof(float);
  __bf16*   Xbf   = (__bf16*)ws;   ws += (size_t)N * KPAD * sizeof(__bf16);
  __bf16*   Ebf   = (__bf16*)ws;   ws += (size_t)E * KPAD * sizeof(__bf16);
  __bf16*   WT4   = (__bf16*)ws;   ws += (size_t)4 * KPAD * OUTF * sizeof(__bf16);
  __bf16*   WeT   = (__bf16*)ws;   ws += (size_t)KPAD * OUTF * sizeof(__bf16);

  const long nEH = (long)E * 4;
  const long nEC = (long)E * OUTF;
  const long nNC = (long)N * OUTF;
  const long nXE = (long)N * KPAD;

  // One-time: gather + convert edge features (shared by both layers)
  pack_edge_kernel<<<(int)((nEC + 255) / 256), 256, 0, stream>>>(ea, emb, Ebf, nEC);

  for (int l = 0; l < 2; ++l) {
    const float* X = (l == 0) ? x : Hbuf;
    const int    F = (l == 0) ? INF_ : OUTF;
    float* target  = (l == 0) ? Hbuf : (float*)d_out;
    const int relu = (l == 0) ? 1 : 0;

    hipMemsetAsync(accum, 0, (size_t)N * OUTF * sizeof(float), stream);
    hipMemsetAsync(mmax,  0, (size_t)N * 4 * sizeof(unsigned), stream);
    hipMemsetAsync(ssum,  0, (size_t)N * 4 * sizeof(float), stream);

    // pack operands for this layer
    pack_x_kernel<<<(int)((nXE + 255) / 256), 256, 0, stream>>>(X, Xbf, nXE, F);
    pack_wT_kernel<<<(KPAD * OUTF + 255) / 256, 256, 0, stream>>>(W[l][0], WT4 + 0 * KPAD * OUTF, F);
    pack_wT_kernel<<<(KPAD * OUTF + 255) / 256, 256, 0, stream>>>(W[l][1], WT4 + 1 * KPAD * OUTF, F);
    pack_wT_kernel<<<(KPAD * OUTF + 255) / 256, 256, 0, stream>>>(W[l][2], WT4 + 2 * KPAD * OUTF, F);
    pack_wT_kernel<<<(KPAD * OUTF + 255) / 256, 256, 0, stream>>>(W[l][4], WT4 + 3 * KPAD * OUTF, F);
    pack_wT_kernel<<<(KPAD * OUTF + 255) / 256, 256, 0, stream>>>(W[l][3], WeT, EDIMK);

    node_gemm_kernel<<<N / 16, 128, 0, stream>>>(
        Xbf, WT4, B[l][0], B[l][1], B[l][2], B[l][3], QKVS);

    edge_gemm_kernel<<<E / 64, 128, 0, stream>>>(Ebf, WeT, EE);

    edge_alpha_kernel<<<(int)((nEH + 255) / 256), 256, 0, stream>>>(
        QKVS, EE, src, dst, alpha, mmax, nEH);

    edge_expsum_kernel<<<(int)((nEH + 255) / 256), 256, 0, stream>>>(
        dst, mmax, alpha, ssum, nEH);

    edge_scatter_kernel<<<(int)((nEC + 255) / 256), 256, 0, stream>>>(
        QKVS, EE, src, dst, alpha, ssum, accum, nEC);

    finalize_kernel<<<(int)((nNC + 255) / 256), 256, 0, stream>>>(
        accum, QKVS, target, nNC, relu);
  }
}
